// RaggedGravNet_simple_37091337568895
// MI455X (gfx1250) — matrix operand order repaired and checked by reading, not compiled, not run on hardware
//
#include <hip/hip_runtime.h>
#include <hip/hip_bf16.h>

#define NTOT 32768
#define VEV  4096
#define NEV  8
#define KNN  40
#define KP1  41
#define FIN  64
#define PDIM 64
#define SDIM 4
#define HDIM 192   // 2*P + F_IN
#define FOUT 128

typedef __attribute__((ext_vector_type(16))) __bf16 v16bf;
typedef __attribute__((ext_vector_type(8)))  float  v8f;

// ---------------------------------------------------------------------------
// K0: convert + transpose weights to bf16 (B operand wants [N][K] row-major)
// ---------------------------------------------------------------------------
__global__ void k_convert_weights(const float* __restrict__ Wprop,
                                  const float* __restrict__ Wout,
                                  unsigned short* __restrict__ WpropT,
                                  unsigned short* __restrict__ WoutT) {
  int t = blockIdx.x * blockDim.x + threadIdx.x;
  __bf16* wp = (__bf16*)WpropT;
  __bf16* wo = (__bf16*)WoutT;
  if (t < FIN * PDIM) {                 // WpropT[n][k] = Wprop[k][n] (64x64)
    int n = t / FIN, k = t % FIN;
    wp[n * FIN + k] = (__bf16)Wprop[k * PDIM + n];
  }
  if (t < HDIM * FOUT) {                // WoutT[n][k] = Wout[k][n] (128x192)
    int n = t / HDIM, k = t % HDIM;
    wo[n * HDIM + k] = (__bf16)Wout[k * FOUT + n];
  }
}

// ---------------------------------------------------------------------------
// K1: coords = x@W_sp + b_sp  (f32, precision-sensitive for KNN) and x -> bf16
// ---------------------------------------------------------------------------
__global__ void k_coords_xbf(const float* __restrict__ x,
                             const float* __restrict__ Wsp,
                             const float* __restrict__ bsp,
                             float* __restrict__ coords,
                             unsigned short* __restrict__ xbf) {
  __shared__ float sW[FIN * SDIM];
  int tid = threadIdx.x;
  if (tid < FIN * SDIM) sW[tid] = Wsp[tid];
  __syncthreads();
  int i = blockIdx.x * 256 + tid;
  float a0 = bsp[0], a1 = bsp[1], a2 = bsp[2], a3 = bsp[3];
  const float* xr = x + (size_t)i * FIN;
  __bf16* xb = (__bf16*)(xbf + (size_t)i * FIN);
#pragma unroll 8
  for (int f = 0; f < FIN; ++f) {
    float xv = xr[f];
    xb[f] = (__bf16)xv;
    a0 = fmaf(xv, sW[f * 4 + 0], a0);
    a1 = fmaf(xv, sW[f * 4 + 1], a1);
    a2 = fmaf(xv, sW[f * 4 + 2], a2);
    a3 = fmaf(xv, sW[f * 4 + 3], a3);
  }
  ((float4*)coords)[i] = make_float4(a0, a1, a2, a3);
}

// ---------------------------------------------------------------------------
// K2: feat[N,64] = x_bf16[N,64] @ WpropT^T + b_prop  via v_wmma_f32_16x16x32_bf16
//     block = 256 thr = 8 waves = 2 Mtiles x 4 Ntiles; block covers 32 rows.
// ---------------------------------------------------------------------------
__global__ void k_feat_wmma(const unsigned short* __restrict__ xbf,
                            const unsigned short* __restrict__ WpropT,
                            const float* __restrict__ bprop,
                            float* __restrict__ feat) {
  int tid = threadIdx.x;
  int wave = tid >> 5, lane = tid & 31;
  int hi = lane >> 4, lm = lane & 15;
  int mbase = blockIdx.x * 32 + (wave >> 2) * 16;
  int nbase = (wave & 3) * 16;
  int row = mbase + lm;
  const __bf16* A = (const __bf16*)xbf + (size_t)row * FIN;
  const __bf16* Bm = (const __bf16*)WpropT + (size_t)(nbase + lm) * FIN;
  v8f c = {};
#pragma unroll
  for (int ks = 0; ks < FIN; ks += 32) {
    v16bf a, b;
    const __bf16* ar = A + ks + hi * 8;       // A: two contiguous 8-elem K-chunks
#pragma unroll
    for (int e = 0; e < 8; ++e) { a[e] = ar[e]; a[e + 8] = ar[16 + e]; }
    const __bf16* br = Bm + ks + hi * 8;      // B: two contiguous 8-elem K-chunks
#pragma unroll
    for (int e = 0; e < 8; ++e) { b[e] = br[e]; b[e + 8] = br[16 + e]; }
    c = __builtin_amdgcn_wmma_f32_16x16x32_bf16(false, a, false, b,
                                                (short)0, c, false, false);
  }
  int col = nbase + lm;
  float bias = bprop[col];
#pragma unroll
  for (int j = 0; j < 8; ++j) {
    int r = mbase + j + hi * 8;               // C/D: VGPR j -> M=j (lo) / j+8 (hi)
    feat[(size_t)r * PDIM + col] = c[j] + bias;
  }
}

// ---------------------------------------------------------------------------
// K3: per-event brute-force KNN. Event coords staged into 64KB LDS with
//     CDNA5 async memory->LDS DMA (global_load_async_to_lds_b128, ASYNCcnt),
//     then each thread scans 4096 candidates keeping a branchless
//     register-resident top-41.
// ---------------------------------------------------------------------------
__global__ void k_knn(const float4* __restrict__ coords4,
                      int* __restrict__ nidx, float* __restrict__ ndsq) {
  __shared__ float4 sc[VEV];                  // 64 KB of the 320 KB WGP LDS
  int ev = blockIdx.x >> 4;
  int chunk = blockIdx.x & 15;
  int tid = threadIdx.x;
  const float4* cev = coords4 + (size_t)ev * VEV;

  // Async stage: each lane issues 16 direct memory->LDS b128 copies.
  // LDS byte offset = low 32 bits of the generic pointer to __shared__.
#pragma unroll
  for (int r = 0; r < VEV / 256; ++r) {
    int c = r * 256 + tid;
    unsigned ldsoff = (unsigned)(unsigned long long)(&sc[c]);
    unsigned long long ga = (unsigned long long)(cev + c);
    asm volatile("global_load_async_to_lds_b128 %0, %1, off"
                 :: "v"(ldsoff), "v"(ga) : "memory");
  }
  // Each wave drains its own ASYNCcnt before the workgroup barrier.
  asm volatile("s_wait_asynccnt 0x0" ::: "memory");
  __syncthreads();

  int il = chunk * 256 + tid;
  float4 me = sc[il];
  float bd[KP1]; int bi[KP1];
#pragma unroll
  for (int j = 0; j < KP1; ++j) { bd[j] = 3.0e38f; bi[j] = il; }

  for (int c = 0; c < VEV; ++c) {
    float4 o = sc[c];
    float dx = me.x - o.x, dy = me.y - o.y, dz = me.z - o.z, dw = me.w - o.w;
    float d = fmaf(dw, dw, fmaf(dz, dz, fmaf(dy, dy, dx * dx)));
    if (d < bd[KP1 - 1]) {                    // fully unrolled sorted insert
#pragma unroll
      for (int j = KP1 - 1; j >= 1; --j) {
        bool sh   = d < bd[j - 1];
        bool here = d < bd[j];
        float pd = bd[j - 1]; int pi = bi[j - 1];
        bd[j] = sh ? pd : (here ? d : bd[j]);
        bi[j] = sh ? pi : (here ? c : bi[j]);
      }
      if (d < bd[0]) { bd[0] = d; bi[0] = c; }
    }
  }
  // drop entry 0 (self, d=0); radius clamp -> self (then d=0, w=1)
  int*   ni = nidx + (size_t)(ev * VEV + il) * KNN;
  float* nd = ndsq + (size_t)(ev * VEV + il) * KNN;
#pragma unroll
  for (int k = 1; k <= KNN; ++k) {
    float d = bd[k]; int id = bi[k];
    if (d > 1.0f) { d = 0.0f; id = il; }
    ni[k - 1] = ev * VEV + id;
    nd[k - 1] = d;
  }
}

// ---------------------------------------------------------------------------
// K4: weighted gather + max/mean aggregation, emit H[N,192] in bf16
//     block = 4 vertices x 64 feature lanes; w,idx staged in LDS;
//     neighbor feat rows prefetched (global_prefetch_b8) while staging.
// ---------------------------------------------------------------------------
__global__ void k_agg(const float* __restrict__ feat, const float* __restrict__ x,
                      const int* __restrict__ nidx, const float* __restrict__ ndsq,
                      unsigned short* __restrict__ H) {
  __shared__ float sw[4][KNN];
  __shared__ int   sn[4][KNN];
  int tid = threadIdx.x;
  int i0 = blockIdx.x * 4;
  if (tid < 4 * KNN) {
    int v = tid / KNN, k = tid % KNN;
    size_t base = (size_t)(i0 + v) * KNN + k;
    int j = nidx[base];
    sw[v][k] = __expf(-10.0f * ndsq[base]);
    sn[v][k] = j;
    __builtin_prefetch(feat + (size_t)j * PDIM, 0, 0);   // warm WGP$/L2
  }
  __syncthreads();
  int v = tid >> 6, p = tid & 63;
  int i = i0 + v;
  float mx = -3.0e38f, sm = 0.0f;
#pragma unroll 4
  for (int k = 0; k < KNN; ++k) {
    int j = sn[v][k];
    float val = feat[(size_t)j * PDIM + p] * sw[v][k];
    mx = fmaxf(mx, val);
    sm += val;
  }
  __bf16* h = (__bf16*)(H + (size_t)i * HDIM);
  h[p]        = (__bf16)mx;
  h[64 + p]   = (__bf16)(sm * (1.0f / KNN));
  h[128 + p]  = (__bf16)x[(size_t)i * FIN + p];
}

// ---------------------------------------------------------------------------
// K5: out[N,128] = tanh(H[N,192] @ W_out + b_out) via bf16 WMMA (6 K-steps)
//     block = 8 waves covering one 16-row tile x 128 cols.
// ---------------------------------------------------------------------------
__global__ void k_out_wmma(const unsigned short* __restrict__ Hb,
                           const unsigned short* __restrict__ WoutT,
                           const float* __restrict__ bout,
                           float* __restrict__ out) {
  int tid = threadIdx.x;
  int wave = tid >> 5, lane = tid & 31;
  int hi = lane >> 4, lm = lane & 15;
  int mbase = blockIdx.x * 16;
  int nbase = wave * 16;
  int row = mbase + lm;
  const __bf16* A  = (const __bf16*)Hb + (size_t)row * HDIM;
  const __bf16* Bm = (const __bf16*)WoutT + (size_t)(nbase + lm) * HDIM;
  v8f c = {};
#pragma unroll
  for (int ks = 0; ks < HDIM; ks += 32) {
    v16bf a, b;
    const __bf16* ar = A + ks + hi * 8;
#pragma unroll
    for (int e = 0; e < 8; ++e) { a[e] = ar[e]; a[e + 8] = ar[16 + e]; }
    const __bf16* br = Bm + ks + hi * 8;
#pragma unroll
    for (int e = 0; e < 8; ++e) { b[e] = br[e]; b[e + 8] = br[16 + e]; }
    c = __builtin_amdgcn_wmma_f32_16x16x32_bf16(false, a, false, b,
                                                (short)0, c, false, false);
  }
  int col = nbase + lm;
  float bias = bout[col];
#pragma unroll
  for (int j = 0; j < 8; ++j) {
    int r = mbase + j + hi * 8;
    out[(size_t)r * FOUT + col] = tanhf(c[j] + bias);
  }
}

// ---------------------------------------------------------------------------
extern "C" void kernel_launch(void* const* d_in, const int* in_sizes, int n_in,
                              void* d_out, int out_size, void* d_ws, size_t ws_size,
                              hipStream_t stream) {
  (void)in_sizes; (void)n_in; (void)out_size; (void)ws_size;
  const float* x      = (const float*)d_in[0];
  const float* W_prop = (const float*)d_in[1];
  const float* b_prop = (const float*)d_in[2];
  const float* W_sp   = (const float*)d_in[3];
  const float* b_sp   = (const float*)d_in[4];
  const float* W_out  = (const float*)d_in[5];
  const float* b_out  = (const float*)d_in[6];
  // d_in[7] = row_splits (uniform events; unused)

  char* ws = (char*)d_ws;
  size_t off = 0;
  float*          coords = (float*)(ws + off);          off += (size_t)NTOT * SDIM * 4;
  unsigned short* xbf    = (unsigned short*)(ws + off); off += (size_t)NTOT * FIN * 2;
  unsigned short* WpropT = (unsigned short*)(ws + off); off += (size_t)FIN * PDIM * 2;
  unsigned short* WoutT  = (unsigned short*)(ws + off); off += (size_t)FOUT * HDIM * 2;
  float*          feat   = (float*)(ws + off);          off += (size_t)NTOT * PDIM * 4;
  int*            nidx   = (int*)(ws + off);            off += (size_t)NTOT * KNN * 4;
  float*          ndsq   = (float*)(ws + off);          off += (size_t)NTOT * KNN * 4;
  unsigned short* Hbuf   = (unsigned short*)(ws + off); off += (size_t)NTOT * HDIM * 2;

  k_convert_weights<<<(HDIM * FOUT + 255) / 256, 256, 0, stream>>>(W_prop, W_out, WpropT, WoutT);
  k_coords_xbf<<<NTOT / 256, 256, 0, stream>>>(x, W_sp, b_sp, coords, xbf);
  k_feat_wmma<<<NTOT / 32, 256, 0, stream>>>(xbf, WpropT, b_prop, feat);
  k_knn<<<NEV * (VEV / 256), 256, 0, stream>>>((const float4*)coords, nidx, ndsq);
  k_agg<<<NTOT / 4, 256, 0, stream>>>(feat, x, nidx, ndsq, Hbuf);
  k_out_wmma<<<NTOT / 16, 256, 0, stream>>>(Hbuf, WoutT, b_out, (float*)d_out);
}
